// SimpleTransformer_4715874091910
// MI455X (gfx1250) — matrix operand order repaired and checked by reading, not compiled, not run on hardware
//
#include <hip/hip_runtime.h>

typedef _Float16 v16h __attribute__((ext_vector_type(16)));
typedef _Float16 v8h  __attribute__((ext_vector_type(8)));
typedef float    v8f  __attribute__((ext_vector_type(8)));
typedef int      v4i  __attribute__((vector_size(16)));

typedef __attribute__((address_space(1))) v4i* as1_v4i_p;
typedef __attribute__((address_space(3))) v4i* as3_v4i_p;

constexpr int Bn   = 4;
constexpr int Tn   = 4096;
constexpr int Dn   = 512;
constexpr int En   = 3 * Dn;   // 1536
constexpr int HIST = 128;
constexpr int KT   = 160;      // padded key window (144 valid + 16 zero pad)

union F16x16 { v16h v; v8h h[2]; };

// ------------------------------------------------ CDNA5 async copy to LDS
__device__ __forceinline__ void async_cp_b128(void* lds_dst, const void* gsrc) {
#if __has_builtin(__builtin_amdgcn_global_load_async_to_lds_b128)
  __builtin_amdgcn_global_load_async_to_lds_b128(
      (as1_v4i_p)(void*)gsrc, (as3_v4i_p)lds_dst, 0, 0);
#else
  __attribute__((address_space(3))) char* l =
      (__attribute__((address_space(3))) char*)lds_dst;
  unsigned loff = (unsigned)(unsigned long long)l;
  asm volatile("global_load_async_to_lds_b128 %0, %1, off"
               :: "v"(loff), "v"(gsrc) : "memory");
#endif
}

__device__ __forceinline__ void wait_async0() {
#if __has_builtin(__builtin_amdgcn_s_wait_asynccnt)
  __builtin_amdgcn_s_wait_asynccnt(0);
#else
  asm volatile("s_wait_asynccnt 0x0" ::: "memory");
#endif
}

// ---------------------------------------------------------------- f32 -> f16
__global__ void cvt_f32_f16(const float* __restrict__ src,
                            _Float16* __restrict__ dst, int n) {
  int i = blockIdx.x * blockDim.x + threadIdx.x;
  if (i < n) dst[i] = (_Float16)src[i];
}

// ------------------------------------------------------- QKV projection GEMM
// qkv[m][e] = sum_d x[m][d] * W[e][d] + bias[e]
// M = B*T = 16384, N = 1536, K = 512.
// Block = 8 waves; each wave owns a 16(M) x 64(N) tile. Grid = (128, 24).
// The block's 64 W rows are staged in LDS via async copies, double-buffered
// in K-chunks of 128 (padded stride -> conflict-free ds_read_b128).
// Output: q_h[m][d], k_h[m][d] row-major f16; v stored transposed v_t[b][d][t].
__global__ void qkv_gemm(const _Float16* __restrict__ xh,
                         const _Float16* __restrict__ wh,
                         const float* __restrict__ bias,
                         _Float16* __restrict__ qh,
                         _Float16* __restrict__ kh,
                         _Float16* __restrict__ vt) {
  constexpr int CK  = 128;        // K chunk staged per step
  constexpr int NC  = Dn / CK;    // 4 chunks
  constexpr int SWS = CK + 8;     // padded LDS row stride in halves (272 B)
  __shared__ _Float16 sW[2][64 * SWS];

  const int tid  = threadIdx.x;
  const int lane = tid & 31;
  const int wave = tid >> 5;
  const int lrow = lane & 15;
  const int lhi  = lane >> 4;

  const int m0 = blockIdx.x * 128 + wave * 16;
  const int e0 = blockIdx.y * 64;

  // staging assignment: 4 threads per W row, 64 B (4 x b128) each
  const int srow  = tid >> 2;
  const int squad = tid & 3;

  auto stage = [&](int buf, int c) {
    const _Float16* src = wh + (e0 + srow) * Dn + c * CK + squad * 32;
    _Float16* dst = &sW[buf][srow * SWS + squad * 32];
#pragma unroll
    for (int j = 0; j < 4; ++j)
      async_cp_b128(dst + j * 8, src + j * 8);
  };

  v8f acc[4] = {};
  const _Float16* xrow = xh + (m0 + lrow) * Dn + (lhi << 3);

  stage(0, 0);
  wait_async0();
  __syncthreads();

  for (int c = 0; c < NC; ++c) {
    if (c + 1 < NC) stage((c + 1) & 1, c + 1);
    const _Float16* wbuf = sW[c & 1];
#pragma unroll
    for (int kk = 0; kk < CK / 32; ++kk) {
      const int k0 = c * CK + kk * 32;
      F16x16 a;
      a.h[0] = *(const v8h*)(xrow + k0);        // K = k0+base .. +7
      a.h[1] = *(const v8h*)(xrow + k0 + 16);   // K = k0+base+16 .. +23
#pragma unroll
      for (int nt = 0; nt < 4; ++nt) {
        const _Float16* wr = wbuf + (nt * 16 + lrow) * SWS + kk * 32 + (lhi << 3);
        F16x16 b;
        b.h[0] = *(const v8h*)(wr);             // ds_read_b128
        b.h[1] = *(const v8h*)(wr + 16);
        acc[nt] = __builtin_amdgcn_wmma_f32_16x16x32_f16(
            false, a.v, false, b.v, (short)0, acc[nt], false, false);
      }
    }
    wait_async0();     // next chunk's copies have landed
    __syncthreads();   // everyone done reading current buffer
  }

#pragma unroll
  for (int nt = 0; nt < 4; ++nt) {
    const int e    = e0 + nt * 16 + lrow;   // column of this lane
    const float bv = bias[e];
    const int seg  = e >> 9;                // 0=q 1=k 2=v
    const int d    = e & (Dn - 1);
#pragma unroll
    for (int i = 0; i < 8; ++i) {
      const int m = m0 + i + (lhi << 3);    // row
      const _Float16 hv = (_Float16)(acc[nt][i] + bv);
      if (seg == 0) {
        qh[m * Dn + d] = hv;
      } else if (seg == 1) {
        kh[m * Dn + d] = hv;
      } else {
        const int bb = m >> 12;             // m / Tn
        const int t  = m & (Tn - 1);
        vt[(bb * Dn + d) * Tn + t] = hv;    // transposed store
      }
    }
  }
}

// ------------------------------------------------------------ band attention
// One block (9 waves, 288 threads) per 16-query tile of one batch.
// Keys in [t0-128, t0+15] -> 9 key tiles of 16.
__global__ void attn_kernel(const _Float16* __restrict__ qh,
                            const _Float16* __restrict__ kh,
                            const _Float16* __restrict__ vt,
                            float* __restrict__ out) {
  __shared__ float    sS[16 * KT];
  __shared__ _Float16 sP[16 * KT];

  const int lane = threadIdx.x & 31;
  const int wave = threadIdx.x >> 5;   // 0..8
  const int lrow = lane & 15;
  const int lhi  = lane >> 4;

  const int nq    = Tn / 16;                      // 256 query tiles per batch
  const int b     = blockIdx.x / nq;
  const int t0    = (blockIdx.x % nq) * 16;
  const int kbase = t0 - HIST;

  // ---- Phase 1: S = Q K^T ; one 16x16 score tile per wave
  {
    v8f acc = {};
    const _Float16* qrow = qh + (b * Tn + t0 + lrow) * Dn + (lhi << 3);
    int j = kbase + wave * 16 + lrow;             // key row for this lane
    if (j < 0) j = 0;                             // clamp; masked in softmax
    const _Float16* krow = kh + (b * Tn + j) * Dn + (lhi << 3);
    for (int k0 = 0; k0 < Dn; k0 += 32) {
      F16x16 a, bb;
      a.h[0]  = *(const v8h*)(qrow + k0);
      a.h[1]  = *(const v8h*)(qrow + k0 + 16);
      bb.h[0] = *(const v8h*)(krow + k0);
      bb.h[1] = *(const v8h*)(krow + k0 + 16);
      acc = __builtin_amdgcn_wmma_f32_16x16x32_f16(
          false, a.v, false, bb.v, (short)0, acc, false, false);
    }
#pragma unroll
    for (int i = 0; i < 8; ++i)
      sS[(i + (lhi << 3)) * KT + wave * 16 + lrow] = acc[i];
  }
  __syncthreads();

  // ---- Phase 2a: masked softmax, one thread per query row
  if (threadIdx.x < 16) {
    const int r = threadIdx.x;
    const float scale = 0.0441941738241592f;      // 1/sqrt(512)
    float mx = -3.0e38f;
    for (int c = r; c <= r + HIST; ++c) {         // valid: r <= c <= r+128
      if (kbase + c >= 0) {
        float v = sS[r * KT + c] * scale;
        mx = fmaxf(mx, v);
      }
    }
    float sum = 0.f;
    for (int c = 0; c < KT; ++c) {
      float p = 0.f;
      if (c >= r && c <= r + HIST && kbase + c >= 0) {
        p = __expf(sS[r * KT + c] * scale - mx);
        sum += p;
      }
      sS[r * KT + c] = p;
    }
    const float inv = 1.f / sum;
    for (int c = 0; c < KT; ++c)
      sP[r * KT + c] = (_Float16)(sS[r * KT + c] * inv);
  }
  __syncthreads();

  // ---- Phase 2b: O = P V ; waves split the 32 output d-tiles
  for (int nt = wave; nt < 32; nt += 9) {
    const int d0 = nt * 16;
    v8f acc = {};
    const _Float16* vbase = vt + (b * Dn + d0 + lrow) * Tn;
#pragma unroll
    for (int kk = 0; kk < 5; ++kk) {
      const int c0 = kk * 32;
      F16x16 a, bb;
      const _Float16* prow = &sP[lrow * KT + c0 + (lhi << 4)];
      a.h[0] = *(const v8h*)(prow);
      a.h[1] = *(const v8h*)(prow + 8);
      int jb = kbase + c0 + (lhi << 4);
      if (jb < 0) jb = 0;                         // P==0 there -> contributes 0
      if (jb > Tn - 16) jb = Tn - 16;             // pad cols have P==0
      bb.h[0] = *(const v8h*)(vbase + jb);
      bb.h[1] = *(const v8h*)(vbase + jb + 8);
      acc = __builtin_amdgcn_wmma_f32_16x16x32_f16(
          false, a.v, false, bb.v, (short)0, acc, false, false);
    }
    float* orow = out + (b * Tn + t0) * Dn + d0 + lrow;
#pragma unroll
    for (int i = 0; i < 8; ++i)
      orow[(i + (lhi << 3)) * Dn] = acc[i];
  }
}

// ---------------------------------------------------------------------------
extern "C" void kernel_launch(void* const* d_in, const int* in_sizes, int n_in,
                              void* d_out, int out_size, void* d_ws, size_t ws_size,
                              hipStream_t stream) {
  const float* x    = (const float*)d_in[0];   // (B, T, D)
  const float* W    = (const float*)d_in[1];   // (3D, D)
  const float* bias = (const float*)d_in[2];   // (3D,)
  float* out = (float*)d_out;                  // (B, T, D) f32

  char* ws = (char*)d_ws;
  size_t off = 0;
  _Float16* xh = (_Float16*)(ws + off); off += (size_t)Bn * Tn * Dn * 2;
  _Float16* wh = (_Float16*)(ws + off); off += (size_t)En * Dn * 2;
  _Float16* qh = (_Float16*)(ws + off); off += (size_t)Bn * Tn * Dn * 2;
  _Float16* kh = (_Float16*)(ws + off); off += (size_t)Bn * Tn * Dn * 2;
  _Float16* vt = (_Float16*)(ws + off); off += (size_t)Bn * Tn * Dn * 2;

  const int nx = Bn * Tn * Dn;
  const int nw = En * Dn;
  cvt_f32_f16<<<(nx + 255) / 256, 256, 0, stream>>>(x, xh, nx);
  cvt_f32_f16<<<(nw + 255) / 256, 256, 0, stream>>>(W, wh, nw);

  dim3 gA(Bn * Tn / 128, En / 64);
  qkv_gemm<<<gA, 256, 0, stream>>>(xh, wh, bias, qh, kh, vt);

  attn_kernel<<<Bn * (Tn / 16), 288, 0, stream>>>(qh, kh, vt, out);
}
